// CocoNODE_20736102105356
// MI455X (gfx1250) — compile-verified, matrix operand order checked by reading, not run on hardware
//
#include <hip/hip_runtime.h>
#include <cstdint>
#include <cstddef>

// ---------------- CDNA5 WMMA / TDM types ----------------
typedef __bf16 bf16;
typedef __attribute__((ext_vector_type(16))) __bf16 v16bf;
typedef __attribute__((ext_vector_type(8)))  __bf16 v8bf;
typedef __attribute__((ext_vector_type(8)))  float  v8f;
typedef __attribute__((ext_vector_type(4)))  unsigned int v4u;
typedef __attribute__((ext_vector_type(8)))  int v8i;
typedef __attribute__((ext_vector_type(4)))  int v4i;

__device__ __forceinline__ v8f wmma_bf16(v16bf a, v16bf b, v8f c) {
  // D = A(16x32 bf16) * B(32x16 bf16) + C(16x16 f32)
  return __builtin_amdgcn_wmma_f32_16x16x32_bf16(false, a, false, b, (short)0, c,
                                                 false, false);
}

// Load a 16x32 bf16 A/B fragment for one lane: contiguous K runs
// [kb, kb+8) and [kb+16, kb+24) relative to the row base (kb = half*8).
__device__ __forceinline__ v16bf load_frag(const bf16* p) {
  v16bf r;
  *((v8bf*)&r)     = *(const v8bf*)(p);
  *((v8bf*)&r + 1) = *(const v8bf*)(p + 16);
  return r;
}

__device__ __forceinline__ float gelu_exact(float x) {
  return 0.5f * x * (1.0f + erff(x * 0.70710678118654752f));
}

// ---------------- model constants ----------------
#define BB   2
#define TT   1024
#define MM   (BB * TT)      // 2048 rows
#define DD   768
#define NHH  12
#define HDD  64
#define TEE  64
#define DTE  (DD + TEE)     // 832
#define VV   50304
#define NBLK 4
#define NSTP 4
#define HSTEP 0.25f

// ---------------- generic WMMA GEMM ----------------
// C[M,N] = op(A[M,K] @ B + bias)
// flags: bit0 GELU epilogue
//        bit1 B is fp32 [N,K] row-major (pre-transposed), cooperative staging
//        bit2 residual: C[m,n] += scale*(acc+bias)
//        bit3 B is bf16 [N,K] row-major; stage 128x32 tiles via Tensor Data Mover
#define GF_GELU  1
#define GF_BT    2
#define GF_RES   4
#define GF_TDM   8

__global__ __launch_bounds__(256) void gemm_wmma(
    const float* __restrict__ A, const void* __restrict__ Bv,
    const float* __restrict__ bias, float* __restrict__ C,
    int M, int N, int K, int flags, float scale)
{
  constexpr int TK = 32;
  constexpr int LP = 8;                     // LDS row pad (keeps rows 16B-aligned)
  __shared__ __align__(16) bf16 As[128][TK + LP];
  __shared__ __align__(16) bf16 Bs[128][TK + LP];

  const float* Bf = (const float*)Bv;       // GF_BT / default path
  const bf16*  Bh = (const bf16*)Bv;        // GF_TDM path

  const int tid  = threadIdx.x;
  const int wid  = tid >> 5;
  const int lane = tid & 31;
  const int half = lane >> 4;
  const int l16  = lane & 15;
  const int wm   = wid & 3;                 // wave M index 0..3 (32 rows each)
  const int wn   = wid >> 2;                // wave N index 0..1 (64 cols each)
  const int m0   = blockIdx.y * 128;
  const int n0   = blockIdx.x * 128;
  const bool bT   = (flags & GF_BT)  != 0;
  const bool bTdm = (flags & GF_TDM) != 0;

  v8f acc[2][4];
#pragma unroll
  for (int mi = 0; mi < 2; ++mi)
#pragma unroll
    for (int ni = 0; ni < 4; ++ni) acc[mi][ni] = (v8f){0,0,0,0,0,0,0,0};

  const int arow = tid >> 1;                // 0..127
  const int akp  = (tid & 1) * 16;          // 0 / 16
  const int bkk  = tid >> 3;                // 0..31  (non-transposed B)
  const int bnp  = (tid & 7) * 16;          // 0..112
  const uint32_t ldsB = (uint32_t)(uintptr_t)&Bs[0][0];  // LDS offset = addr[31:0]

  for (int k0 = 0; k0 < K; k0 += TK) {
    // ---- stage A tile (fp32 -> bf16) ----
    {
      const float4* src = (const float4*)(A + (size_t)(m0 + arow) * K + k0 + akp);
      bf16* dst = &As[arow][akp];
#pragma unroll
      for (int i = 0; i < 4; ++i) {
        float4 f = src[i];
        dst[i*4+0] = (bf16)f.x; dst[i*4+1] = (bf16)f.y;
        dst[i*4+2] = (bf16)f.z; dst[i*4+3] = (bf16)f.w;
      }
    }
    // ---- stage B tile as [n][k] ----
    if (bTdm) {
      // Tensor Data Mover: DMA a 128(n) x 32(k) bf16 tile of B[N,K] into Bs,
      // inserting LP(=8 bf16 = 4 DWORDs) of LDS padding every 16 DWORDs (one row).
      if (wid == 0) {
        const uint64_t ga = (uint64_t)(uintptr_t)Bh +
                            (((uint64_t)n0 * (uint64_t)K + (uint64_t)k0) << 1);
        v4u g0;
        g0.x = 1u;                                    // count=1, user descriptor
        g0.y = ldsB;                                  // lds_addr
        g0.z = (uint32_t)ga;                          // global_addr[31:0]
        g0.w = (uint32_t)((ga >> 32) & 0x1FFFFFFu) | (2u << 30);  // addr[56:32], type=2
        // group1: data_size=2B; pad_enable, pad_interval=3 (16 DW), pad_amount=3 (4 DW)
        const uint32_t w0 = (1u << 16) | (1u << 20) | (3u << 22) | (3u << 25);
        const uint32_t w1 = ((uint32_t)K & 0xFFFFu) << 16;                 // tdim0 lo
        const uint32_t w2 = ((uint32_t)K >> 16) | (((uint32_t)N & 0xFFFFu) << 16);
        const uint32_t w3 = ((uint32_t)N >> 16) | (32u << 16);             // tile_dim0=32
        const uint32_t w4 = 128u;                                          // tile_dim1=128
        const uint32_t w5 = (uint32_t)K;                                   // dim0_stride
        v8i g1;
        g1[0] = (int)w0; g1[1] = (int)w1; g1[2] = (int)w2; g1[3] = (int)w3;
        g1[4] = (int)w4; g1[5] = (int)w5; g1[6] = 0; g1[7] = 0;
        v4i zz4 = (v4i){0,0,0,0};
#if defined(__clang_major__) && (__clang_major__ >= 23)
        v8i zz8 = (v8i){0,0,0,0,0,0,0,0};
        __builtin_amdgcn_tensor_load_to_lds(g0, g1, zz4, zz4, zz8, 0);
#else
        __builtin_amdgcn_tensor_load_to_lds(g0, g1, zz4, zz4, 0);
#endif
        __builtin_amdgcn_s_wait_tensorcnt(0);
      }
    } else if (bT) {
      const float4* src = (const float4*)(Bf + (size_t)(n0 + arow) * K + k0 + akp);
      bf16* dst = &Bs[arow][akp];
#pragma unroll
      for (int i = 0; i < 4; ++i) {
        float4 f = src[i];
        dst[i*4+0] = (bf16)f.x; dst[i*4+1] = (bf16)f.y;
        dst[i*4+2] = (bf16)f.z; dst[i*4+3] = (bf16)f.w;
      }
    } else {
      const float4* src = (const float4*)(Bf + (size_t)(k0 + bkk) * N + n0 + bnp);
#pragma unroll
      for (int i = 0; i < 4; ++i) {
        float4 f = src[i];
        Bs[bnp + i*4 + 0][bkk] = (bf16)f.x;
        Bs[bnp + i*4 + 1][bkk] = (bf16)f.y;
        Bs[bnp + i*4 + 2][bkk] = (bf16)f.z;
        Bs[bnp + i*4 + 3][bkk] = (bf16)f.w;
      }
    }
    // prefetch next K tile while this one computes (global_prefetch_b8)
    if (k0 + TK < K) {
      __builtin_prefetch(A + (size_t)(m0 + arow) * K + (k0 + TK) + akp, 0, 1);
      if (bT) __builtin_prefetch(Bf + (size_t)(n0 + arow) * K + (k0 + TK) + akp, 0, 1);
      else if (!bTdm)
        __builtin_prefetch(Bf + (size_t)(k0 + TK + bkk) * N + n0 + bnp, 0, 1);
    }
    __syncthreads();

    const int kb = half * 8;
    v16bf afr[2], bfr[4];
#pragma unroll
    for (int mi = 0; mi < 2; ++mi)
      afr[mi] = load_frag(&As[wm*32 + mi*16 + l16][kb]);
#pragma unroll
    for (int ni = 0; ni < 4; ++ni)
      bfr[ni] = load_frag(&Bs[wn*64 + ni*16 + l16][kb]);
#pragma unroll
    for (int mi = 0; mi < 2; ++mi)
#pragma unroll
      for (int ni = 0; ni < 4; ++ni)
        acc[mi][ni] = wmma_bf16(afr[mi], bfr[ni], acc[mi][ni]);
    __syncthreads();
  }

  // ---- epilogue: bias / GELU / residual ----
  const bool doG = (flags & GF_GELU) != 0;
  const bool doR = (flags & GF_RES)  != 0;
#pragma unroll
  for (int mi = 0; mi < 2; ++mi) {
#pragma unroll
    for (int ni = 0; ni < 4; ++ni) {
      const int n = n0 + wn*64 + ni*16 + l16;
      const float bv = bias ? bias[n] : 0.0f;
#pragma unroll
      for (int r = 0; r < 8; ++r) {
        const int m = m0 + wm*32 + mi*16 + half*8 + r;   // C-layout: M = r + half*8
        float v = acc[mi][ni][r] + bv;
        if (doG) v = gelu_exact(v);
        const size_t o = (size_t)m * N + n;
        if (doR) C[o] += scale * v; else C[o] = v;
      }
    }
  }
}

// ---------------- weight convert+transpose: fp32 [K,N] -> bf16 [N,K] ----------------
__global__ __launch_bounds__(256) void convT(const float* __restrict__ in,
                                             bf16* __restrict__ out, int K, int N)
{
  const size_t i = (size_t)blockIdx.x * 256 + threadIdx.x;   // over N*K
  const int n = (int)(i / K), k = (int)(i % K);
  out[i] = (bf16)in[(size_t)k * N + n];
}

// ---------------- flash-style causal attention (WMMA) ----------------
// grid: (T/64, NH, B), block 128 (4 waves, 16 queries each)
__global__ __launch_bounds__(128) void attn_wmma(
    const float* __restrict__ qkv, float* __restrict__ y)
{
  __shared__ __align__(16) bf16 Qs[64][72];     // [q][d], pre-scaled by 1/8
  __shared__ __align__(16) bf16 Ks[16][72];     // [key][d]
  __shared__ __align__(16) bf16 Vt[64][40];     // [d][key], keys 16..31 == 0
  __shared__ __align__(16) bf16 Ps[4][16][40];  // per-wave P [q][key], keys 16..31 == 0

  const int tid  = threadIdx.x;
  const int wid  = tid >> 5;
  const int lane = tid & 31;
  const int half = lane >> 4;
  const int l16  = lane & 15;
  const int qb   = blockIdx.x * 64;
  const int hH   = blockIdx.y;
  const int bb   = blockIdx.z;
  const size_t rs = 3 * DD;                      // qkv row stride
  const float* base = qkv + (size_t)bb * TT * rs;

  // zero the padded key columns (written once, never overwritten)
  for (int i = tid; i < 64*40; i += 128) {
    int d = i / 40, c = i % 40;
    if (c >= 16) Vt[d][c] = (bf16)0.0f;
  }
  for (int i = tid; i < 4*16*40; i += 128) {
    int w = i / 640, rr = (i % 640) / 40, c = i % 40;
    if (c >= 16) Ps[w][rr][c] = (bf16)0.0f;
  }
  // load + scale Q block
  {
    const int row = tid >> 1, dp = (tid & 1) * 32;
    const float* src = base + (size_t)(qb + row) * rs + hH*HDD + dp;
#pragma unroll
    for (int i = 0; i < 32; ++i) Qs[row][dp + i] = (bf16)(src[i] * 0.125f);
  }
  __syncthreads();

  const int kb = half * 8;
  v16bf qfr[2];
#pragma unroll
  for (int ks = 0; ks < 2; ++ks)
    qfr[ks] = load_frag(&Qs[wid*16 + l16][ks*32 + kb]);

  float mi_[8], li_[8];
  v8f accO[4];
#pragma unroll
  for (int r = 0; r < 8; ++r) { mi_[r] = -1e30f; li_[r] = 0.0f; }
#pragma unroll
  for (int ni = 0; ni < 4; ++ni) accO[ni] = (v8f){0,0,0,0,0,0,0,0};

  const int nj = qb / 16 + 4;                    // key blocks with any unmasked entry
  for (int j = 0; j < nj; ++j) {
    __syncthreads();                             // protect Ks/Vt readers of prev iter
    {
      const int row = tid >> 3, dp = (tid & 7) * 8;
      const float* sk = base + (size_t)(j*16 + row) * rs + DD     + hH*HDD + dp;
      const float* sv = base + (size_t)(j*16 + row) * rs + 2*DD   + hH*HDD + dp;
#pragma unroll
      for (int i = 0; i < 8; ++i) Ks[row][dp + i] = (bf16)sk[i];
#pragma unroll
      for (int i = 0; i < 8; ++i) Vt[dp + i][row] = (bf16)sv[i];
    }
    __syncthreads();

    // S = (Q/8) @ K^T  (16q x 16k)
    v8f s = (v8f){0,0,0,0,0,0,0,0};
#pragma unroll
    for (int ks = 0; ks < 2; ++ks) {
      v16bf kf = load_frag(&Ks[l16][ks*32 + kb]);
      s = wmma_bf16(qfr[ks], kf, s);
    }

    // online softmax (rows live across 16 lanes of one half)
#pragma unroll
    for (int r = 0; r < 8; ++r) {
      const int qg = qb + wid*16 + half*8 + r;
      const int kg = j*16 + l16;
      float sv = (kg <= qg) ? s[r] : -1e30f;
      float mx = sv;
      for (int o = 8; o; o >>= 1) mx = fmaxf(mx, __shfl_xor(mx, o));
      const float mnew = fmaxf(mi_[r], mx);
      const float p = __expf(sv - mnew);
      float sm = p;
      for (int o = 8; o; o >>= 1) sm += __shfl_xor(sm, o);
      const float corr = __expf(mi_[r] - mnew);
      li_[r] = li_[r] * corr + sm;
      mi_[r] = mnew;
#pragma unroll
      for (int ni = 0; ni < 4; ++ni) accO[ni][r] *= corr;
      Ps[wid][half*8 + r][l16] = (bf16)p;        // C-layout -> LDS
    }
    // P (A-layout, keys 16..31 zero) @ V  -> O accum
    v16bf pf = load_frag(&Ps[wid][l16][kb]);
#pragma unroll
    for (int ni = 0; ni < 4; ++ni) {
      v16bf vf = load_frag(&Vt[ni*16 + l16][kb]);
      accO[ni] = wmma_bf16(pf, vf, accO[ni]);
    }
  }

  // normalize + store y[b, q, h*64 + d]
#pragma unroll
  for (int ni = 0; ni < 4; ++ni) {
#pragma unroll
    for (int r = 0; r < 8; ++r) {
      const int qg = qb + wid*16 + half*8 + r;
      const int d  = hH*HDD + ni*16 + l16;
      y[(size_t)(bb*TT + qg) * DD + d] = accO[ni][r] / li_[r];
    }
  }
}

// ---------------- LN with optional time-emb concat ----------------
__global__ __launch_bounds__(256) void ln_concat(
    const float* __restrict__ in, const float* __restrict__ temb,
    const float* __restrict__ w, const float* __restrict__ b,
    float* __restrict__ out, int Din, int TEin)
{
  const int row = blockIdx.x;
  const int Dt  = Din + TEin;
  __shared__ float red[256];
  __shared__ float s_mean, s_inv;
  const float* ir = in + (size_t)row * Din;

  float s = 0.0f;
  for (int c = threadIdx.x; c < Dt; c += 256)
    s += (c < Din) ? ir[c] : temb[c - Din];
  red[threadIdx.x] = s; __syncthreads();
  for (int o = 128; o; o >>= 1) {
    if (threadIdx.x < o) red[threadIdx.x] += red[threadIdx.x + o];
    __syncthreads();
  }
  if (threadIdx.x == 0) s_mean = red[0] / (float)Dt;
  __syncthreads();
  const float mean = s_mean;

  s = 0.0f;
  for (int c = threadIdx.x; c < Dt; c += 256) {
    float v = (c < Din) ? ir[c] : temb[c - Din];
    float d = v - mean; s += d * d;
  }
  red[threadIdx.x] = s; __syncthreads();
  for (int o = 128; o; o >>= 1) {
    if (threadIdx.x < o) red[threadIdx.x] += red[threadIdx.x + o];
    __syncthreads();
  }
  if (threadIdx.x == 0) s_inv = rsqrtf(red[0] / (float)Dt + 1e-5f);
  __syncthreads();
  const float inv = s_inv;

  float* orow = out + (size_t)row * Dt;
  for (int c = threadIdx.x; c < Dt; c += 256) {
    float v = (c < Din) ? ir[c] : temb[c - Din];
    orow[c] = (v - mean) * inv * w[c] + b[c];
  }
}

// ---------------- tiny time MLP: temb = gelu(t*w1+b1) @ w2 + b2 ----------------
__global__ void time_mlp(const float* __restrict__ w1, const float* __restrict__ b1,
                         const float* __restrict__ w2, const float* __restrict__ b2,
                         float t, float* __restrict__ temb)
{
  __shared__ float h[TEE];
  const int n = threadIdx.x;
  h[n] = gelu_exact(t * w1[n] + b1[n]);
  __syncthreads();
  float s = b2[n];
#pragma unroll 8
  for (int j = 0; j < TEE; ++j) s += h[j] * w2[j * TEE + n];
  temb[n] = s;
}

// ---------------- embedding gather ----------------
__global__ void embed(const int* __restrict__ idx, const float* __restrict__ wte,
                      const float* __restrict__ wpe, float* __restrict__ z)
{
  const size_t i = (size_t)blockIdx.x * blockDim.x + threadIdx.x;  // MM*DD total
  const int m = (int)(i / DD), d = (int)(i % DD);
  const int t = m % TT;
  z[i] = wte[(size_t)idx[m] * DD + d] + wpe[(size_t)t * DD + d];
}

// ---------------- host launch ----------------
extern "C" void kernel_launch(void* const* d_in, const int* in_sizes, int n_in,
                              void* d_out, int out_size, void* d_ws, size_t ws_size,
                              hipStream_t stream)
{
  (void)in_sizes; (void)n_in; (void)out_size; (void)ws_size;
  const int*   idx         = (const int*)  d_in[0];
  const float* wte         = (const float*)d_in[1];
  const float* wpe         = (const float*)d_in[2];
  const float* time_w1     = (const float*)d_in[3];
  const float* time_b1     = (const float*)d_in[4];
  const float* time_w2     = (const float*)d_in[5];
  const float* time_b2     = (const float*)d_in[6];
  const float* attn_ln_w   = (const float*)d_in[7];
  const float* attn_ln_b   = (const float*)d_in[8];
  const float* c_attn_w    = (const float*)d_in[9];
  const float* c_attn_b    = (const float*)d_in[10];
  const float* attn_proj_w = (const float*)d_in[11];
  const float* attn_proj_b = (const float*)d_in[12];
  const float* mlp_ln_w    = (const float*)d_in[13];
  const float* mlp_ln_b    = (const float*)d_in[14];
  const float* c_fc_w      = (const float*)d_in[15];
  const float* c_fc_b      = (const float*)d_in[16];
  const float* mlp_proj_w  = (const float*)d_in[17];
  const float* mlp_proj_b  = (const float*)d_in[18];
  const float* ln_f_w      = (const float*)d_in[19];
  const float* ln_f_b      = (const float*)d_in[20];

  // workspace layout: ~76 MB fp32 activations + ~59 MB bf16 weights
  float* z    = (float*)d_ws;                 // [MM, DD]
  float* x    = z    + (size_t)MM * DD;       // [MM, DTE]
  float* qkv  = x    + (size_t)MM * DTE;      // [MM, 3*DD]
  float* y    = qkv  + (size_t)MM * 3 * DD;   // [MM, DD]
  float* h    = y    + (size_t)MM * DD;       // [MM, DD]
  float* fc   = h    + (size_t)MM * DD;       // [MM, 4*DD]
  float* zf   = fc   + (size_t)MM * 4 * DD;   // [MM, DD]
  float* temb = zf   + (size_t)MM * DD;       // [TEE]
  bf16*  wbf  = (bf16*)(temb + TEE);          // bf16 transposed weights

  const size_t SZ_CA = (size_t)(3*DD) * DTE;      // c_attn^T   [2304, 832]
  const size_t SZ_AP = (size_t)DD * DD;           // attn_proj^T [768, 768]
  const size_t SZ_CF = (size_t)(4*DD) * DTE;      // c_fc^T     [3072, 832]
  const size_t SZ_MP = (size_t)DD * (4*DD);       // mlp_proj^T  [768, 3072]
  const size_t BLK_STRIDE = SZ_CA + SZ_AP + SZ_CF + SZ_MP;

  // one-time (per launch) weight convert + transpose to bf16 [N,K]
  for (int blk = 0; blk < NBLK; ++blk) {
    bf16* wb = wbf + (size_t)blk * BLK_STRIDE;
    convT<<<(unsigned)(SZ_CA / 256), 256, 0, stream>>>(
        c_attn_w + (size_t)blk * SZ_CA, wb, DTE, 3*DD);
    convT<<<(unsigned)(SZ_AP / 256), 256, 0, stream>>>(
        attn_proj_w + (size_t)blk * SZ_AP, wb + SZ_CA, DD, DD);
    convT<<<(unsigned)(SZ_CF / 256), 256, 0, stream>>>(
        c_fc_w + (size_t)blk * SZ_CF, wb + SZ_CA + SZ_AP, DTE, 4*DD);
    convT<<<(unsigned)(SZ_MP / 256), 256, 0, stream>>>(
        mlp_proj_w + (size_t)blk * SZ_MP, wb + SZ_CA + SZ_AP + SZ_CF, 4*DD, DD);
  }

  embed<<<(MM * DD) / 256, 256, 0, stream>>>(idx, wte, wpe, z);

  for (int blk = 0; blk < NBLK; ++blk) {
    const float* tw1 = time_w1     + (size_t)blk * TEE;
    const float* tb1 = time_b1     + (size_t)blk * TEE;
    const float* tw2 = time_w2     + (size_t)blk * TEE * TEE;
    const float* tb2 = time_b2     + (size_t)blk * TEE;
    const float* alw = attn_ln_w   + (size_t)blk * DTE;
    const float* alb = attn_ln_b   + (size_t)blk * DTE;
    const float* cab = c_attn_b    + (size_t)blk * 3 * DD;
    const float* apb = attn_proj_b + (size_t)blk * DD;
    const float* mlw = mlp_ln_w    + (size_t)blk * DTE;
    const float* mlb = mlp_ln_b    + (size_t)blk * DTE;
    const float* cfb = c_fc_b      + (size_t)blk * 4 * DD;
    const float* mpb = mlp_proj_b  + (size_t)blk * DD;
    const bf16*  wb  = wbf + (size_t)blk * BLK_STRIDE;
    const bf16*  caw_t = wb;
    const bf16*  apw_t = wb + SZ_CA;
    const bf16*  cfw_t = wb + SZ_CA + SZ_AP;
    const bf16*  mpw_t = wb + SZ_CA + SZ_AP + SZ_CF;

    for (int st = 0; st < NSTP; ++st) {
      const float t = HSTEP * (float)st;
      time_mlp<<<1, TEE, 0, stream>>>(tw1, tb1, tw2, tb2, t, temb);
      // x = LN(cat(z, temb))
      ln_concat<<<MM, 256, 0, stream>>>(z, temb, alw, alb, x, DD, TEE);
      // qkv = x @ c_attn_w + b          (TDM-staged bf16 weights)
      gemm_wmma<<<dim3(3*DD/128, MM/128), 256, 0, stream>>>(
          x, caw_t, cab, qkv, MM, 3*DD, DTE, GF_TDM, 0.0f);
      // y = causal attention
      attn_wmma<<<dim3(TT/64, NHH, BB), 128, 0, stream>>>(qkv, y);
      // h = y @ attn_proj_w + b
      gemm_wmma<<<dim3(DD/128, MM/128), 256, 0, stream>>>(
          y, apw_t, apb, h, MM, DD, DD, GF_TDM, 0.0f);
      // x = LN(cat(h, temb))
      ln_concat<<<MM, 256, 0, stream>>>(h, temb, mlw, mlb, x, DD, TEE);
      // fc = gelu(x @ c_fc_w + b)
      gemm_wmma<<<dim3(4*DD/128, MM/128), 256, 0, stream>>>(
          x, cfw_t, cfb, fc, MM, 4*DD, DTE, GF_GELU | GF_TDM, 0.0f);
      // z += 0.25 * (fc @ mlp_proj_w + b)
      gemm_wmma<<<dim3(DD/128, MM/128), 256, 0, stream>>>(
          fc, mpw_t, mpb, z, MM, DD, 4*DD, GF_RES | GF_TDM, HSTEP);
    }
  }

  // zf = LN_f(z); logits = zf @ wte^T  (wte is fp32 [V, D] == B[N,K] row-major)
  ln_concat<<<MM, 256, 0, stream>>>(z, nullptr, ln_f_w, ln_f_b, zf, DD, 0);
  gemm_wmma<<<dim3(VV/128, MM/128), 256, 0, stream>>>(
      zf, wte, nullptr, (float*)d_out, MM, VV, DD, GF_BT, 0.0f);
}
